// ImgExemplarSelfAttn_54219667145029
// MI455X (gfx1250) — matrix operand order repaired and checked by reading, not compiled, not run on hardware
//
#include <hip/hip_runtime.h>
#include <cstdint>
#include <cstddef>

#define B_    8
#define C_    256
#define HW_   4096
#define NTOT_ 4097
#define NPAD_ 4160   // 130*32 = 65*64
#define TOPK_ 150

typedef __bf16 bf16;
typedef __attribute__((ext_vector_type(16))) bf16  v16bf;
typedef __attribute__((ext_vector_type(8)))  bf16  v8bf;
typedef __attribute__((ext_vector_type(8)))  float v8f;

typedef __attribute__((ext_vector_type(4))) unsigned int u32x4;
typedef __attribute__((ext_vector_type(8))) int          i32x8;
typedef __attribute__((ext_vector_type(4))) int          i32x4;

#if __has_builtin(__builtin_amdgcn_tensor_load_to_lds)
#define HAVE_TDM 1
#else
#define HAVE_TDM 0
#endif

// D = A(16x32 bf16) x B(32x16 bf16) + C(16x16 f32)
__device__ __forceinline__ v8f wmma_bf16(v16bf a, v16bf b, v8f c) {
  return __builtin_amdgcn_wmma_f32_16x16x32_bf16(false, a, false, b, (short)0, c,
                                                 false, false);
}

// Gather this lane's A-fragment (16x32) from a row-major bf16 row pointer.
// Layout: lanes 0-15 hold K={0..15}, lanes 16-31 K={16..31} (matches load_bfrag).
__device__ __forceinline__ v16bf load_afrag(const bf16* rowp, int k0, int hf) {
  v16bf f;
  const bf16* q = rowp + k0 + hf * 16;
#pragma unroll
  for (int jj = 0; jj < 8; ++jj) {
    int K = jj << 1;
    f[2 * jj]     = q[K];
    f[2 * jj + 1] = q[K + 1];
  }
  return f;
}

// B-fragment: 16 contiguous bf16 (k-major row per output column), as 2x b128.
__device__ __forceinline__ v16bf load_bfrag(const bf16* p) {
  v8bf lo = *(const v8bf*)p;
  v8bf hi = *(const v8bf*)(p + 8);
  v16bf f;
#pragma unroll
  for (int j = 0; j < 8; ++j) { f[j] = lo[j]; f[j + 8] = hi[j]; }
  return f;
}

#if HAVE_TDM
// Issue a TDM 2D tile load: tile_d1 rows of tile_d0 elements (2-byte), global row
// stride = stride_elems, LDS destination padded per-row via pad codes.
// clang-23 builtin: (u32x4 g0, i32x8 g1, i32x4 g2, i32x4 g3, i32x8 g4, i32 cpol)
__device__ __forceinline__ void tdm_load_2d(unsigned int lds_off, const void* gptr,
                                            unsigned int tile_d0, unsigned int tile_d1,
                                            unsigned int tensor_d0, unsigned int tensor_d1,
                                            unsigned int stride_elems,
                                            unsigned int pad_interval_code,
                                            unsigned int pad_amount_code) {
  unsigned long long ga = (unsigned long long)(uintptr_t)gptr;
  u32x4 g0;
  g0[0] = 1u;                                         // count=1, user mode, no gather
  g0[1] = lds_off;                                    // lds_addr
  g0[2] = (unsigned int)ga;                           // global_addr[31:0]
  g0[3] = (unsigned int)((ga >> 32) & 0x1FFFFFFull)   // global_addr[56:32]
          | (2u << 30);                               // type = 2 ("image")
  i32x8 g1;
  g1[0] = (int)((1u << 16)                            // data_size = 2 bytes
                | (1u << 20)                          // pad_enable
                | (pad_interval_code << 22) | (pad_amount_code << 25));
  g1[1] = (int)((tensor_d0 & 0xFFFFu) << 16);                                 // dim0 lo16
  g1[2] = (int)(((tensor_d0 >> 16) & 0xFFFFu) | ((tensor_d1 & 0xFFFFu) << 16));
  g1[3] = (int)(((tensor_d1 >> 16) & 0xFFFFu) | ((tile_d0 & 0xFFFFu) << 16));
  g1[4] = (int)(tile_d1 & 0xFFFFu);                   // tile_dim1 (tile_dim2 = 0)
  g1[5] = (int)stride_elems;                          // tensor_dim0_stride[31:0]
  g1[6] = 0;
  g1[7] = 0;
  i32x4 g2; g2[0] = 0; g2[1] = 0; g2[2] = 0; g2[3] = 0;
  i32x4 g3; g3[0] = 0; g3[1] = 0; g3[2] = 0; g3[3] = 0;
  i32x8 g4; g4[0] = 0; g4[1] = 0; g4[2] = 0; g4[3] = 0;
  g4[4] = 0; g4[5] = 0; g4[6] = 0; g4[7] = 0;
  __builtin_amdgcn_tensor_load_to_lds(g0, g1, g2, g3, g4, 0);
}
#endif

// ----------------------------------------- prep: build concatT[b][n][c] (bf16, transposed)
__global__ __launch_bounds__(256) void prep_kernel(const float* __restrict__ img,
                                                   const float* __restrict__ exe,
                                                   bf16* __restrict__ concatT) {
  __shared__ bf16 t[64 * 72];  // [cc][nn]
  const int b  = blockIdx.x;
  const int n0 = blockIdx.y * 64;   // 65 tiles, last covers 4096..4159
  const int c0 = blockIdx.z * 64;
  const int tid = threadIdx.x;
  for (int i = tid; i < 64 * 64; i += 256) {
    int cc = i >> 6, nn = i & 63;
    int n = n0 + nn;
    float v;
    if (n < HW_)       v = img[((size_t)b * C_ + c0 + cc) * HW_ + n];
    else if (n == HW_) v = 1.2f * exe[b * C_ + c0 + cc];
    else               v = 0.f;
    t[cc * 72 + nn] = (bf16)v;
  }
  __syncthreads();
  for (int i = tid; i < 64 * 64; i += 256) {
    int nn = i >> 6, cc = i & 63;
    concatT[((size_t)b * NPAD_ + n0 + nn) * C_ + c0 + cc] = t[cc * 72 + nn];
  }
}

// zero padded tail rows of values_bf16 (rows NTOT_..NPAD_-1)
__global__ void ztail_kernel(bf16* __restrict__ valsB) {
  int per = (NPAD_ - NTOT_) * C_;
  int i = blockIdx.x * blockDim.x + threadIdx.x;
  if (i >= B_ * per) return;
  int b = i / per, rem = i % per;
  int row = NTOT_ + rem / C_, d = rem % C_;
  valsB[((size_t)b * NPAD_ + row) * C_ + d] = (bf16)0.f;
}

// ------------------------------------------------- GEMM for values (mode 0) / proj (mode 1)
// mode 0: out[b][n][d] = sum_c A[n][c] * W[d][c]   (A = concat^T, incl. exe row tile)
// mode 1: out[b][m][d] = sum_c A[m][c] * W[c][d]   (A = unrolled^T, W = similarity)
__global__ __launch_bounds__(128) void gemm_kernel(
    const float* __restrict__ img, const float* __restrict__ exe,
    const float* __restrict__ W, float* __restrict__ outF,
    bf16* __restrict__ outB, int mode) {
  __shared__ bf16 sA[32 * 264];  // [row][c]
  __shared__ bf16 sB[64 * 264];  // [dcol][c]  (k-major per output column)
  const int b = blockIdx.x;
  const int n0 = blockIdx.y * 32;
  const int d0 = blockIdx.z * 64;
  const int tid = threadIdx.x;
  const int wv = tid >> 5, lane = tid & 31, hf = lane >> 4, l16 = lane & 15;

  if (mode == 0 && blockIdx.y == 128) {          // exe row tile (row 4096)
    for (int i = tid; i < 32 * 256; i += 128) {
      int nl = i >> 8, c = i & 255;
      float v = (nl == 0) ? 1.2f * exe[b * C_ + c] : 0.f;
      sA[nl * 264 + c] = (bf16)v;
    }
  } else {
    const float* ip = img + (size_t)b * C_ * HW_ + n0;
    for (int i = tid; i < C_ * 32; i += 128) {   // coalesced over n, transpose into LDS
      int c = i >> 5, nl = i & 31;
      sA[nl * 264 + c] = (bf16)ip[(size_t)c * HW_ + nl];
    }
  }
  if (mode == 0) {                                // B[c][d] = W[d][c] -> store rows of W
    for (int i = tid; i < 64 * 256; i += 128) {
      int dc = i >> 8, c = i & 255;
      sB[dc * 264 + c] = (bf16)W[(size_t)(d0 + dc) * C_ + c];
    }
  } else {                                        // B[c][d] = W[c][d] -> transpose
    for (int i = tid; i < C_ * 64; i += 128) {
      int c = i >> 6, dc = i & 63;
      sB[dc * 264 + c] = (bf16)W[(size_t)c * C_ + d0 + dc];
    }
  }
  __syncthreads();

  const int rowbase = (wv & 1) * 16;
  const int colbase = (wv >> 1) * 32;
  v8f acc[2];
#pragma unroll
  for (int s = 0; s < 2; ++s)
#pragma unroll
    for (int r = 0; r < 8; ++r) acc[s][r] = 0.f;

#pragma unroll
  for (int kc = 0; kc < 8; ++kc) {
    v16bf a = load_afrag(&sA[(rowbase + l16) * 264], kc * 32, hf);
#pragma unroll
    for (int s = 0; s < 2; ++s) {
      v16bf bfr = load_bfrag(&sB[(colbase + s * 16 + l16) * 264 + kc * 32 + hf * 16]);
      acc[s] = wmma_bf16(a, bfr, acc[s]);
    }
  }

  const size_t rows = (mode == 0) ? (size_t)NPAD_ : (size_t)HW_;
#pragma unroll
  for (int s = 0; s < 2; ++s)
#pragma unroll
    for (int r = 0; r < 8; ++r) {
      int grow = n0 + rowbase + r + 8 * hf;
      int gcol = d0 + colbase + s * 16 + l16;
      size_t idx = ((size_t)b * rows + grow) * C_ + gcol;
      float v = acc[s][r];
      if (mode == 0) { outF[idx] = v; outB[idx] = (bf16)v; }
      else           { outB[idx] = (bf16)v; }
    }
}

// ---------------------------------- values transpose: valsT[b][d][n] from valsB[b][n][d]
__global__ __launch_bounds__(256) void vtrans_kernel(const bf16* __restrict__ valsB,
                                                     bf16* __restrict__ valsT) {
  __shared__ bf16 t[64 * 72];  // [dd][nn]
  const int b  = blockIdx.x;
  const int n0 = blockIdx.y * 64;
  const int d0 = blockIdx.z * 64;
  const int tid = threadIdx.x;
  for (int i = tid; i < 64 * 64; i += 256) {
    int nn = i >> 6, dd = i & 63;
    t[dd * 72 + nn] = valsB[((size_t)b * NPAD_ + n0 + nn) * C_ + d0 + dd];
  }
  __syncthreads();
  for (int i = tid; i < 64 * 64; i += 256) {
    int dd = i >> 6, nn = i & 63;
    valsT[((size_t)b * C_ + d0 + dd) * NPAD_ + n0 + nn] = t[dd * 72 + nn];
  }
}

// ------------------------------------------------- exe-column attention scores
__global__ void scores_kernel(const bf16* __restrict__ projB, const float* __restrict__ exe,
                              float* __restrict__ scores) {
  int t = blockIdx.x * blockDim.x + threadIdx.x;
  if (t >= B_ * HW_) return;
  int b = t / HW_;
  const bf16* pr = projB + (size_t)t * C_;
  const float* ex = exe + b * C_;
  float s = 0.f;
  for (int d = 0; d < C_; ++d) s += (float)pr[d] * (1.2f * ex[d]);
  scores[t] = s;
}

// ------------------------------------------------- top-150 per batch (argmax loop, low-index ties)
__global__ __launch_bounds__(256) void topk_kernel(const float* __restrict__ scores,
                                                   int* __restrict__ qids) {
  __shared__ float buf[HW_];
  __shared__ float rv[256];
  __shared__ int   ri[256];
  int b = blockIdx.x, t = threadIdx.x;
  for (int i = t; i < HW_; i += 256) buf[i] = scores[(size_t)b * HW_ + i];
  __syncthreads();
  for (int k = 0; k < TOPK_; ++k) {
    float bv = -3.0e38f; int bi = HW_;
    for (int i = t; i < HW_; i += 256) {
      float v = buf[i];
      if (v > bv) { bv = v; bi = i; }   // ascending i => strict '>' keeps lowest index
    }
    rv[t] = bv; ri[t] = bi;
    __syncthreads();
    for (int o = 128; o > 0; o >>= 1) {
      if (t < o) {
        float v2 = rv[t + o]; int i2 = ri[t + o];
        if (v2 > rv[t] || (v2 == rv[t] && i2 < ri[t])) { rv[t] = v2; ri[t] = i2; }
      }
      __syncthreads();
    }
    if (t == 0) { qids[b * TOPK_ + k] = ri[0]; buf[ri[0]] = -3.0e38f; }
    __syncthreads();
  }
}

// ------------------------------------------------- gather queries (f32 values rows)
__global__ void queries_kernel(const float* __restrict__ valsF, const int* __restrict__ qids,
                               float* __restrict__ outq) {
  int i = blockIdx.x * blockDim.x + threadIdx.x;
  if (i >= B_ * TOPK_ * C_) return;
  int d = i & 255;
  int bk = i >> 8;
  int k = bk % TOPK_, b = bk / TOPK_;
  int n = qids[b * TOPK_ + k];
  outq[i] = valsF[((size_t)b * NPAD_ + n) * C_ + d];
}

// ------------------------------------------------- fused flash attention (TDM double-buffered)
// out[b][m][d] = softmax_n(proj[b][m][:]·concat[:][n]) · values[n][d]
#define BT_ELEMS  (32 * 264)                 // concat tile  [n_local][c], stride 264
#define VT_ELEMS  (256 * 40)                 // values tile  [d][n_local], stride 40
#define BUF_ELEMS (BT_ELEMS + VT_ELEMS)
#define PL_ELEMS  (4 * 16 * 40)
#define SMEM_BYTES ((2 * BUF_ELEMS + PL_ELEMS) * 2)

__global__ __launch_bounds__(128) void flash_kernel(
    const bf16* __restrict__ concatT,  // [B][NPAD][C]
    const bf16* __restrict__ projB,    // [B][HW][C]
    const bf16* __restrict__ valsT,    // [B][C][NPAD]
    float* __restrict__ out) {         // [B][HW][C]
  extern __shared__ bf16 smem[];

  const int b   = blockIdx.x;
  const int m0  = blockIdx.y * 64;
  const int tid = threadIdx.x;
  const int wv  = tid >> 5, lane = tid & 31, hf = lane >> 4, l16 = lane & 15;

  // Hold this wave's 16x256 proj slice as 8 A-fragments in registers (reused 129x).
  v16bf APF[8];
  {
    const bf16* rowp = projB + ((size_t)b * HW_ + m0 + wv * 16 + l16) * C_;
#pragma unroll
    for (int kc = 0; kc < 8; ++kc) APF[kc] = load_afrag(rowp, kc * 32, hf);
  }

  v8f O[16];
#pragma unroll
  for (int t = 0; t < 16; ++t)
#pragma unroll
    for (int r = 0; r < 8; ++r) O[t][r] = 0.f;
  float mrow[8], lrow[8];
#pragma unroll
  for (int r = 0; r < 8; ++r) { mrow[r] = -3.0e38f; lrow[r] = 0.f; }

  bf16* pl = smem + 2 * BUF_ELEMS + wv * 16 * 40;

#if HAVE_TDM
  const unsigned int lds_base = (unsigned int)(uintptr_t)(void*)smem;
  // prologue: tile 0 into buffer 0
  if (wv == 0) {
    tdm_load_2d(lds_base, concatT + (size_t)b * NPAD_ * C_,
                /*tile*/ 256, 32, /*tensor*/ 256, NPAD_, /*stride*/ 256,
                /*padi 128 DW*/ 6, /*pada 4 DW*/ 3);
    tdm_load_2d(lds_base + BT_ELEMS * 2, valsT + (size_t)b * C_ * NPAD_,
                /*tile*/ 32, 256, /*tensor*/ NPAD_, C_, /*stride*/ NPAD_,
                /*padi 16 DW*/ 3, /*pada 4 DW*/ 3);
    __builtin_amdgcn_s_wait_tensorcnt(0);
  }
  __syncthreads();
#endif

  for (int nt = 0; nt < 129; ++nt) {   // 129*32 = 4128 >= 4097 valid columns
    const int n0 = nt * 32;
    bf16* sBT = smem + (nt & 1) * BUF_ELEMS;
    bf16* sVT = sBT + BT_ELEMS;

#if HAVE_TDM
    if (wv == 0 && nt + 1 < 129) {     // prefetch next tile into the other buffer
      const int nn0 = n0 + 32;
      unsigned int dst = lds_base + ((nt + 1) & 1) * BUF_ELEMS * 2;
      tdm_load_2d(dst, concatT + ((size_t)b * NPAD_ + nn0) * C_,
                  256, 32, 256, NPAD_, 256, 6, 3);
      tdm_load_2d(dst + BT_ELEMS * 2, valsT + (size_t)b * C_ * NPAD_ + nn0,
                  32, 256, NPAD_, C_, NPAD_, 3, 3);
    }
#else
    __syncthreads();
    {
      const bf16* cb = concatT + ((size_t)b * NPAD_ + n0) * C_;
      for (int i = tid; i < 32 * C_; i += 128) {
        int nl = i >> 8, c = i & 255;
        sBT[nl * 264 + c] = cb[nl * C_ + c];
      }
      const bf16* vb = valsT + (size_t)b * C_ * NPAD_ + n0;
      for (int i = tid; i < C_ * 32; i += 128) {
        int d = i >> 5, nl = i & 31;
        sVT[d * 40 + nl] = vb[(size_t)d * NPAD_ + nl];
      }
    }
    __syncthreads();
#endif

    // S(16x32) = proj_tile x concat_tile
    v8f S[2];
#pragma unroll
    for (int s = 0; s < 2; ++s)
#pragma unroll
      for (int r = 0; r < 8; ++r) S[s][r] = 0.f;
#pragma unroll
    for (int kc = 0; kc < 8; ++kc)
#pragma unroll
      for (int s = 0; s < 2; ++s) {
        v16bf bfr = load_bfrag(&sBT[(s * 16 + l16) * 264 + kc * 32 + hf * 16]);
        S[s] = wmma_bf16(APF[kc], bfr, S[s]);
      }

    // mask columns beyond N=4097
#pragma unroll
    for (int s = 0; s < 2; ++s)
      if (n0 + s * 16 + l16 >= NTOT_) {
#pragma unroll
        for (int r = 0; r < 8; ++r) S[s][r] = -3.0e38f;
      }

    // online softmax: row reductions within 16-lane halves (rows r+8*hf)
    float tmax[8];
#pragma unroll
    for (int r = 0; r < 8; ++r) tmax[r] = fmaxf(S[0][r], S[1][r]);
#pragma unroll
    for (int off = 8; off >= 1; off >>= 1)
#pragma unroll
      for (int r = 0; r < 8; ++r) tmax[r] = fmaxf(tmax[r], __shfl_xor(tmax[r], off, 32));

    float scl[8], tsum[8];
#pragma unroll
    for (int r = 0; r < 8; ++r) {
      float mn = fmaxf(mrow[r], tmax[r]);
      scl[r] = __expf(mrow[r] - mn);
      mrow[r] = mn;
      tsum[r] = 0.f;
    }
#pragma unroll
    for (int s = 0; s < 2; ++s)
#pragma unroll
      for (int r = 0; r < 8; ++r) {
        float p = __expf(S[s][r] - mrow[r]);
        S[s][r] = p;
        tsum[r] += p;
      }
#pragma unroll
    for (int off = 8; off >= 1; off >>= 1)
#pragma unroll
      for (int r = 0; r < 8; ++r) tsum[r] += __shfl_xor(tsum[r], off, 32);
#pragma unroll
    for (int r = 0; r < 8; ++r) lrow[r] = lrow[r] * scl[r] + tsum[r];
#pragma unroll
    for (int t = 0; t < 16; ++t)
#pragma unroll
      for (int r = 0; r < 8; ++r) O[t][r] *= scl[r];

    // P: D-layout -> A-layout via per-wave LDS (DS ops are in-order per wave)
#pragma unroll
    for (int s = 0; s < 2; ++s)
#pragma unroll
      for (int r = 0; r < 8; ++r)
        pl[(r + 8 * hf) * 40 + s * 16 + l16] = (bf16)S[s][r];
    v16bf PA = load_afrag(pl + l16 * 40, 0, hf);

    // O(16x256) += P(16x32) x V(32x256)
#pragma unroll
    for (int t = 0; t < 16; ++t) {
      v16bf vf = load_bfrag(&sVT[(t * 16 + l16) * 40 + hf * 16]);
      O[t] = wmma_bf16(PA, vf, O[t]);
    }

#if HAVE_TDM
    if (wv == 0 && nt + 1 < 129) __builtin_amdgcn_s_wait_tensorcnt(0);
    __syncthreads();   // next buffer visible to all waves; current reads complete
#endif
  }

  float inv[8];
#pragma unroll
  for (int r = 0; r < 8; ++r) inv[r] = 1.0f / lrow[r];
  float* op = out + ((size_t)b * HW_ + m0 + wv * 16) * C_;
#pragma unroll
  for (int t = 0; t < 16; ++t)
#pragma unroll
    for (int r = 0; r < 8; ++r)
      op[(r + 8 * hf) * C_ + t * 16 + l16] = O[t][r] * inv[r];
}

// ----------------------------------------------------------------------------
extern "C" void kernel_launch(void* const* d_in, const int* in_sizes, int n_in,
                              void* d_out, int out_size, void* d_ws, size_t ws_size,
                              hipStream_t stream) {
  const float* img = (const float*)d_in[0];  // (8,256,64,64)
  const float* exe = (const float*)d_in[1];  // (8,256)
  const float* sim = (const float*)d_in[2];  // (256,256)
  const float* vw  = (const float*)d_in[3];  // (256,256)
  float* out   = (float*)d_out;
  float* out_q = out + (size_t)B_ * HW_ * C_;

  char* ws = (char*)d_ws;
  size_t off = 0;
  auto take = [&](size_t bytes) -> char* {
    char* p = ws + off;
    off += (bytes + 255) & ~(size_t)255;
    return p;
  };
  bf16*  concatT = (bf16*)take((size_t)B_ * NPAD_ * C_ * sizeof(bf16));
  bf16*  projB   = (bf16*)take((size_t)B_ * HW_ * C_ * sizeof(bf16));
  bf16*  valsB   = (bf16*)take((size_t)B_ * NPAD_ * C_ * sizeof(bf16));
  bf16*  valsT   = (bf16*)take((size_t)B_ * C_ * NPAD_ * sizeof(bf16));
  float* valsF   = (float*)take((size_t)B_ * NPAD_ * C_ * sizeof(float));
  float* scoresP = (float*)take((size_t)B_ * HW_ * sizeof(float));
  int*   qids    = (int*)take((size_t)B_ * TOPK_ * sizeof(int));
  (void)in_sizes; (void)n_in; (void)out_size; (void)ws_size;

  prep_kernel<<<dim3(B_, 65, 4), 256, 0, stream>>>(img, exe, concatT);
  {
    int n = B_ * (NPAD_ - NTOT_) * C_;
    ztail_kernel<<<(n + 255) / 256, 256, 0, stream>>>(valsB);
  }
  gemm_kernel<<<dim3(B_, 129, 4), 128, 0, stream>>>(img, exe, vw, valsF, valsB, 0);
  vtrans_kernel<<<dim3(B_, 65, 4), 256, 0, stream>>>(valsB, valsT);
  gemm_kernel<<<dim3(B_, 128, 4), 128, 0, stream>>>(img, exe, sim, nullptr, projB, 1);
  scores_kernel<<<(B_ * HW_ + 255) / 256, 256, 0, stream>>>(projB, exe, scoresP);
  topk_kernel<<<B_, 256, 0, stream>>>(scoresP, qids);
  queries_kernel<<<(B_ * TOPK_ * C_ + 255) / 256, 256, 0, stream>>>(valsF, qids, out_q);
  flash_kernel<<<dim3(B_, 64), 128, SMEM_BYTES, stream>>>(concatT, projB, valsT, out);
}
